// SpectralGNN_32744830665390
// MI455X (gfx1250) — matrix operand order repaired and checked by reading, not compiled, not run on hardware
//
#include <hip/hip_runtime.h>
#include <hip/hip_fp16.h>

// ---------------------------------------------------------------------------
// SpectralGNN forward for MI455X (gfx1250, wave32).
// Dense [N,128]x[128,128] GEMMs run on v_wmma_f32_16x16x32_f16 (f16 in / f32
// accum). Weights are staged to LDS via the CDNA5 async global->LDS path
// (global_load_async_to_lds_b128 / s_wait_asynccnt); B-fragments then come
// from ds_load_b128. The three Chebyshev GEMMs of a layer are fused into one
// kernel (96 WMMAs/wave, T2 = 2*spmm - T0 built on the fly), removing the
// accumulator read-back and elementwise passes. Sparse ops (SpMM, pooling,
// Dirichlet reg) use global f32 atomics, L2-resident (h = 25.6 MB << 192 MB).
// ---------------------------------------------------------------------------

typedef __attribute__((ext_vector_type(16))) _Float16 v16h;
typedef __attribute__((ext_vector_type(8)))  float    v8f;
typedef long long i64;

#define N_NODES 50000
#define N_EDGES 800000
#define N_GRAPH 128
#define HDIM    128
#define LLAYERS 3
#define KORD    3
#define EPSV    1e-5f

// ---- async global->LDS staging helpers (CDNA5) ----------------------------
__device__ __forceinline__ void async_cp16(const void* g, unsigned lds_off) {
    asm volatile("global_load_async_to_lds_b128 %0, %1, off"
                 :: "v"(lds_off), "v"((unsigned long long)g) : "memory");
}
__device__ __forceinline__ void async_wait0() {
    asm volatile("s_wait_asynccnt 0x0" ::: "memory");
}

// -------------------------------- utility kernels --------------------------

__global__ void k_zero(float* __restrict__ p, int n) {
    int i = blockIdx.x * 256 + threadIdx.x;
    if (i < n) p[i] = 0.0f;
}

__global__ void k_deg(const i64* __restrict__ src, float* __restrict__ deg, int E) {
    int e = blockIdx.x * 256 + threadIdx.x;
    if (e < E) atomicAdd(deg + (int)src[e], 1.0f);
}

__global__ void k_dinv(const float* __restrict__ deg, float* __restrict__ dinv, int n) {
    int i = blockIdx.x * 256 + threadIdx.x;
    if (i < n) {
        float d = deg[i];
        dinv[i] = (d > 0.0f) ? rsqrtf(fmaxf(d, 1.0f)) : 0.0f;
    }
}

// Convert nm row-major f32 [128x128] weights (k-major) into f16 transposed
// layout Wt[n*128 + k] so WMMA B-fragments are two contiguous 16-byte loads (k and k+16).
__global__ void k_convw(const float* __restrict__ W, _Float16* __restrict__ Wt, int nm) {
    int i = blockIdx.x * 256 + threadIdx.x;
    int tot = nm * HDIM * HDIM;
    if (i < tot) {
        int m = i / (HDIM * HDIM);
        int r = i % (HDIM * HDIM);
        int k = r / HDIM, n = r % HDIM;
        Wt[(size_t)m * HDIM * HDIM + (size_t)n * HDIM + k] =
            (_Float16)W[(size_t)m * HDIM * HDIM + (size_t)k * HDIM + n];
    }
}

// ---- helper: build a 16-bit A-fragment (16x32 layout) from 16 f32 ---------
// p0 points at this lane's first 8-float group; second group at +16 floats.
__device__ __forceinline__ v16h make_afrag(const float* p0) {
    const float4 g0 = *(const float4*)(p0 + 0);
    const float4 g1 = *(const float4*)(p0 + 4);
    const float4 g2 = *(const float4*)(p0 + 16);
    const float4 g3 = *(const float4*)(p0 + 20);
    v16h a;
    a[0]  = (_Float16)g0.x;  a[1]  = (_Float16)g0.y;
    a[2]  = (_Float16)g0.z;  a[3]  = (_Float16)g0.w;
    a[4]  = (_Float16)g1.x;  a[5]  = (_Float16)g1.y;
    a[6]  = (_Float16)g1.z;  a[7]  = (_Float16)g1.w;
    a[8]  = (_Float16)g2.x;  a[9]  = (_Float16)g2.y;
    a[10] = (_Float16)g2.z;  a[11] = (_Float16)g2.w;
    a[12] = (_Float16)g3.x;  a[13] = (_Float16)g3.y;
    a[14] = (_Float16)g3.z;  a[15] = (_Float16)g3.w;
    return a;
}

// -------------------------------- WMMA GEMM (embed) ------------------------
// C[nrows x 128] = A[nrows x 128] @ W[128 x 128] + bias
// Block = 256 threads = 8 waves; each wave owns a 16-row x 128-col strip.
// nrows % 16 == 0 (50000 = 3125*16), so no intra-tile masking is needed.
__global__ __launch_bounds__(256) void k_gemm128(
    const float* __restrict__ A, const _Float16* __restrict__ Wt,
    const float* __restrict__ bias, float* __restrict__ C, int nrows)
{
    __shared__ _Float16 wlds[HDIM * HDIM];   // 32 KB

    const int lane = threadIdx.x & 31;
    const int wv   = threadIdx.x >> 5;
    const int m0   = blockIdx.x * 128 + wv * 16;
    const int half = lane >> 4;
    const int l15  = lane & 15;

    // async-stage the f16 weight tile into LDS (all 256 threads participate)
    {
        auto* l3 = (__attribute__((address_space(3))) _Float16*)wlds;
#pragma unroll
        for (int it = 0; it < (HDIM * HDIM) / (8 * 256); ++it) {
            int idx = (it * 256 + threadIdx.x) * 8;     // halfs (16 B)
            async_cp16(Wt + idx, (unsigned)(unsigned long long)(l3 + idx));
        }
        async_wait0();
    }
    __syncthreads();                   // all waves reach this (no early return)

    if (m0 >= nrows) return;

    v8f acc[8];
#pragma unroll
    for (int t = 0; t < 8; ++t)
#pragma unroll
        for (int r = 0; r < 8; ++r) acc[t][r] = 0.0f;

    // hoist all 4 K-steps of the A fragment into registers
    v16h af[4];
    const float* Arow = A + (size_t)(m0 + l15) * HDIM;
#pragma unroll
    for (int ks = 0; ks < 4; ++ks)
        af[ks] = make_afrag(Arow + ks * 32 + half * 8);

    // 32 WMMAs fed from LDS
#pragma unroll
    for (int ks = 0; ks < 4; ++ks) {
#pragma unroll
        for (int t = 0; t < 8; ++t) {
            const _Float16* bp = &wlds[(t * 16 + l15) * HDIM + ks * 32 + half * 8];
            v16h bf = __builtin_shufflevector(*(const __attribute__((ext_vector_type(8))) _Float16*)bp, *(const __attribute__((ext_vector_type(8))) _Float16*)(bp + 16), 0,1,2,3,4,5,6,7,8,9,10,11,12,13,14,15);
            acc[t] = __builtin_amdgcn_wmma_f32_16x16x32_f16(
                false, af[ks], false, bf, (short)0, acc[t], false, false);
        }
    }

#pragma unroll
    for (int t = 0; t < 8; ++t) {
        const int col = t * 16 + l15;
        const float bv = bias[col];
#pragma unroll
        for (int r = 0; r < 8; ++r)
            C[(size_t)(m0 + half * 8 + r) * HDIM + col] = acc[t][r] + bv;
    }
}

// ----------------------- fused Chebyshev GEMM ------------------------------
// acc = T0@W0 + T1@W1 + T2@W2 + bias, with T0 = h, T1 = t1 (spmm(h)),
// T2 = 2*s - h built on the fly (s = spmm(t1)).  W012 = 3 consecutive
// f16-transposed 128x128 weights; all staged into 96 KB of LDS.
__global__ __launch_bounds__(256) void k_cheb_gemm(
    const float* __restrict__ H, const float* __restrict__ T1,
    const float* __restrict__ S, const _Float16* __restrict__ W012,
    const float* __restrict__ bias, float* __restrict__ C, int nrows)
{
    __shared__ _Float16 wlds[3 * HDIM * HDIM];   // 96 KB of 320 KB/WGP

    const int lane = threadIdx.x & 31;
    const int wv   = threadIdx.x >> 5;
    const int m0   = blockIdx.x * 128 + wv * 16;
    const int half = lane >> 4;
    const int l15  = lane & 15;

    // async-stage all three weight tiles (contiguous in W012)
    {
        auto* l3 = (__attribute__((address_space(3))) _Float16*)wlds;
#pragma unroll
        for (int it = 0; it < (3 * HDIM * HDIM) / (8 * 256); ++it) {  // 24 iters
            int idx = (it * 256 + threadIdx.x) * 8;
            async_cp16(W012 + idx, (unsigned)(unsigned long long)(l3 + idx));
        }
        async_wait0();
    }
    __syncthreads();

    if (m0 >= nrows) return;

    v8f acc[8];
#pragma unroll
    for (int t = 0; t < 8; ++t)
#pragma unroll
        for (int r = 0; r < 8; ++r) acc[t][r] = 0.0f;

    const size_t rowoff = (size_t)(m0 + l15) * HDIM;
    const float* Hrow  = H  + rowoff;
    const float* T1row = T1 + rowoff;
    const float* Srow  = S  + rowoff;

#pragma unroll
    for (int ks = 0; ks < 4; ++ks) {
        const int ko = ks * 32 + half * 8;

        // T0 fragment (also provides the f32 h-values for T2)
        const float4 h0 = *(const float4*)(Hrow + ko + 0);
        const float4 h1 = *(const float4*)(Hrow + ko + 4);
        const float4 h2 = *(const float4*)(Hrow + ko + 16);
        const float4 h3 = *(const float4*)(Hrow + ko + 20);
        v16h af0;
        af0[0]=(_Float16)h0.x; af0[1]=(_Float16)h0.y; af0[2]=(_Float16)h0.z; af0[3]=(_Float16)h0.w;
        af0[4]=(_Float16)h1.x; af0[5]=(_Float16)h1.y; af0[6]=(_Float16)h1.z; af0[7]=(_Float16)h1.w;
        af0[8]=(_Float16)h2.x; af0[9]=(_Float16)h2.y; af0[10]=(_Float16)h2.z; af0[11]=(_Float16)h2.w;
        af0[12]=(_Float16)h3.x; af0[13]=(_Float16)h3.y; af0[14]=(_Float16)h3.z; af0[15]=(_Float16)h3.w;

        // T1 fragment
        v16h af1 = make_afrag(T1row + ko);

        // T2 = 2*s - h fragment (f32 math, then f16 convert)
        const float4 s0 = *(const float4*)(Srow + ko + 0);
        const float4 s1 = *(const float4*)(Srow + ko + 4);
        const float4 s2 = *(const float4*)(Srow + ko + 16);
        const float4 s3 = *(const float4*)(Srow + ko + 20);
        v16h af2;
        af2[0]=(_Float16)(2.0f*s0.x-h0.x); af2[1]=(_Float16)(2.0f*s0.y-h0.y);
        af2[2]=(_Float16)(2.0f*s0.z-h0.z); af2[3]=(_Float16)(2.0f*s0.w-h0.w);
        af2[4]=(_Float16)(2.0f*s1.x-h1.x); af2[5]=(_Float16)(2.0f*s1.y-h1.y);
        af2[6]=(_Float16)(2.0f*s1.z-h1.z); af2[7]=(_Float16)(2.0f*s1.w-h1.w);
        af2[8]=(_Float16)(2.0f*s2.x-h2.x); af2[9]=(_Float16)(2.0f*s2.y-h2.y);
        af2[10]=(_Float16)(2.0f*s2.z-h2.z); af2[11]=(_Float16)(2.0f*s2.w-h2.w);
        af2[12]=(_Float16)(2.0f*s3.x-h3.x); af2[13]=(_Float16)(2.0f*s3.y-h3.y);
        af2[14]=(_Float16)(2.0f*s3.z-h3.z); af2[15]=(_Float16)(2.0f*s3.w-h3.w);

        const int bko = ks * 32 + half * 8;
#pragma unroll
        for (int t = 0; t < 8; ++t) {
            const int boff = (t * 16 + l15) * HDIM + bko;
            v16h b0 = __builtin_shufflevector(*(const __attribute__((ext_vector_type(8))) _Float16*)(&wlds[boff]), *(const __attribute__((ext_vector_type(8))) _Float16*)(&wlds[boff + 16]), 0,1,2,3,4,5,6,7,8,9,10,11,12,13,14,15);
            v16h b1 = __builtin_shufflevector(*(const __attribute__((ext_vector_type(8))) _Float16*)(&wlds[HDIM * HDIM + boff]), *(const __attribute__((ext_vector_type(8))) _Float16*)(&wlds[HDIM * HDIM + boff + 16]), 0,1,2,3,4,5,6,7,8,9,10,11,12,13,14,15);
            v16h b2 = __builtin_shufflevector(*(const __attribute__((ext_vector_type(8))) _Float16*)(&wlds[2 * HDIM * HDIM + boff]), *(const __attribute__((ext_vector_type(8))) _Float16*)(&wlds[2 * HDIM * HDIM + boff + 16]), 0,1,2,3,4,5,6,7,8,9,10,11,12,13,14,15);
            acc[t] = __builtin_amdgcn_wmma_f32_16x16x32_f16(
                false, af0, false, b0, (short)0, acc[t], false, false);
            acc[t] = __builtin_amdgcn_wmma_f32_16x16x32_f16(
                false, af1, false, b1, (short)0, acc[t], false, false);
            acc[t] = __builtin_amdgcn_wmma_f32_16x16x32_f16(
                false, af2, false, b2, (short)0, acc[t], false, false);
        }
    }

#pragma unroll
    for (int t = 0; t < 8; ++t) {
        const int col = t * 16 + l15;
        const float bv = bias[col];
#pragma unroll
        for (int r = 0; r < 8; ++r)
            C[(size_t)(m0 + half * 8 + r) * HDIM + col] = acc[t][r] + bv;
    }
}

// ------------------------- LayerNorm + ReLU (per row) ----------------------
__global__ __launch_bounds__(256) void k_ln_relu(
    const float* __restrict__ X, const float* __restrict__ g,
    const float* __restrict__ b, float* __restrict__ Y, int nrows)
{
    int lane = threadIdx.x & 31;
    int row  = blockIdx.x * 8 + (threadIdx.x >> 5);
    if (row >= nrows) return;
    int c = lane * 4;
    const float4 v = *(const float4*)(X + (size_t)row * HDIM + c);
    float s  = v.x + v.y + v.z + v.w;
    float sq = v.x * v.x + v.y * v.y + v.z * v.z + v.w * v.w;
#pragma unroll
    for (int m = 16; m >= 1; m >>= 1) {
        s  += __shfl_xor(s,  m, 32);
        sq += __shfl_xor(sq, m, 32);
    }
    float mean = s * (1.0f / HDIM);
    float var  = sq * (1.0f / HDIM) - mean * mean;
    float rs   = rsqrtf(var + EPSV);
    float4 o;
    o.x = fmaxf((v.x - mean) * rs * g[c + 0] + b[c + 0], 0.0f);
    o.y = fmaxf((v.y - mean) * rs * g[c + 1] + b[c + 1], 0.0f);
    o.z = fmaxf((v.z - mean) * rs * g[c + 2] + b[c + 2], 0.0f);
    o.w = fmaxf((v.w - mean) * rs * g[c + 3] + b[c + 3], 0.0f);
    *(float4*)(Y + (size_t)row * HDIM + c) = o;
}

// ------------------------- SpMM: y[dst] += w_e * h[src] --------------------
__global__ __launch_bounds__(256) void k_spmm(
    const i64* __restrict__ src, const i64* __restrict__ dst,
    const float* __restrict__ dinv, const float* __restrict__ H,
    float* __restrict__ Y, int E)
{
    long long i = (long long)blockIdx.x * 256 + threadIdx.x;
    int e = (int)(i >> 5);
    if (e >= E) return;
    int c = ((int)i & 31) * 4;
    int s = (int)src[e], d = (int)dst[e];
    float w = -dinv[s] * dinv[d];
    const float4 hv = *(const float4*)(H + (size_t)s * HDIM + c);
    float* yp = Y + (size_t)d * HDIM + c;
    atomicAdd(yp + 0, w * hv.x);
    atomicAdd(yp + 1, w * hv.y);
    atomicAdd(yp + 2, w * hv.z);
    atomicAdd(yp + 3, w * hv.w);
}

// ------------------------- BatchNorm over nodes ----------------------------
__global__ __launch_bounds__(128) void k_colstat(
    const float* __restrict__ X, float* __restrict__ sums, int nrows)
{
    int c  = threadIdx.x;               // one thread per column -> coalesced
    int r0 = blockIdx.x * 256;
    int r1 = min(r0 + 256, nrows);
    float s = 0.0f, q = 0.0f;
    for (int r = r0; r < r1; ++r) {
        float v = X[(size_t)r * HDIM + c];
        s += v; q += v * v;
    }
    atomicAdd(&sums[c], s);
    atomicAdd(&sums[HDIM + c], q);
}

__global__ __launch_bounds__(128) void k_bnfin(
    const float* __restrict__ sums, const float* __restrict__ g,
    const float* __restrict__ b, float* __restrict__ ss, int nrows)
{
    int c = threadIdx.x;
    float inv = 1.0f / (float)nrows;
    float mean = sums[c] * inv;
    float var  = sums[HDIM + c] * inv - mean * mean;
    float sc   = g[c] * rsqrtf(var + EPSV);
    ss[c] = sc;
    ss[HDIM + c] = b[c] - mean * sc;
}

__global__ void k_bnapply(const float* __restrict__ X, const float* __restrict__ ss,
                          float* __restrict__ Y, int n) {
    int i = blockIdx.x * 256 + threadIdx.x;
    if (i < n) {
        int c = i & (HDIM - 1);
        Y[i] = fmaxf(X[i] * ss[c] + ss[HDIM + c], 0.0f);
    }
}

// ------------------------- Dirichlet regularizer ---------------------------
__global__ __launch_bounds__(256) void k_reg(
    const i64* __restrict__ src, const i64* __restrict__ dst,
    const float* __restrict__ H, float* __restrict__ regsum, int E)
{
    __shared__ float red[8];
    long long i = (long long)blockIdx.x * 256 + threadIdx.x;
    int e = (int)(i >> 5);
    float part = 0.0f;
    if (e < E) {
        int c = ((int)i & 31) * 4;
        int s = (int)src[e], d = (int)dst[e];
        const float4 a  = *(const float4*)(H + (size_t)s * HDIM + c);
        const float4 bb = *(const float4*)(H + (size_t)d * HDIM + c);
        float dx = a.x - bb.x, dy = a.y - bb.y, dz = a.z - bb.z, dw = a.w - bb.w;
        part = dx * dx + dy * dy + dz * dz + dw * dw;
    }
#pragma unroll
    for (int m = 16; m >= 1; m >>= 1) part += __shfl_xor(part, m, 32);
    if ((threadIdx.x & 31) == 0) red[threadIdx.x >> 5] = part;
    __syncthreads();
    if (threadIdx.x == 0) {
        float t = 0.0f;
#pragma unroll
        for (int k = 0; k < 8; ++k) t += red[k];
        atomicAdd(regsum, t);
    }
}

// ------------------------- global mean pool + MLP --------------------------
__global__ void k_counts(const i64* __restrict__ batch, float* __restrict__ counts, int n) {
    int i = blockIdx.x * 256 + threadIdx.x;
    if (i < n) atomicAdd(counts + (int)batch[i], 1.0f);
}

__global__ __launch_bounds__(256) void k_gsum(
    const i64* __restrict__ batch, const float* __restrict__ H,
    float* __restrict__ gsum, int n)
{
    long long i = (long long)blockIdx.x * 256 + threadIdx.x;
    int node = (int)(i >> 5);
    if (node >= n) return;
    int c = ((int)i & 31) * 4;
    int g = (int)batch[node];
    const float4 v = *(const float4*)(H + (size_t)node * HDIM + c);
    float* p = gsum + (size_t)g * HDIM + c;
    atomicAdd(p + 0, v.x);
    atomicAdd(p + 1, v.y);
    atomicAdd(p + 2, v.z);
    atomicAdd(p + 3, v.w);
}

__global__ void k_gmean(const float* __restrict__ gsum, const float* __restrict__ counts,
                        float* __restrict__ gmean) {
    int i = blockIdx.x * 256 + threadIdx.x;
    if (i < N_GRAPH * HDIM) {
        int g = i / HDIM;
        gmean[i] = gsum[i] / fmaxf(counts[g], 1.0f);
    }
}

__global__ void k_mlp1(const float* __restrict__ gmean, const float* __restrict__ w1,
                       const float* __restrict__ b1, float* __restrict__ z) {
    int i = blockIdx.x * 256 + threadIdx.x;
    if (i >= N_GRAPH * 64) return;
    int g = i / 64, j = i % 64;
    float s = b1[j];
    for (int c = 0; c < HDIM; ++c) s += gmean[(size_t)g * HDIM + c] * w1[(size_t)c * 64 + j];
    z[i] = fmaxf(s, 0.0f);
}

__global__ void k_mlp2(const float* __restrict__ z, const float* __restrict__ w2,
                       const float* __restrict__ b2, float* __restrict__ out) {
    int i = blockIdx.x * 256 + threadIdx.x;
    if (i >= N_GRAPH * 2) return;
    int g = i / 2, o = i % 2;
    float s = b2[o];
    for (int j = 0; j < 64; ++j) s += z[(size_t)g * 64 + j] * w2[(size_t)j * 2 + o];
    out[i] = s;
}

__global__ void k_regfin(const float* __restrict__ regsum, float* __restrict__ out) {
    if (blockIdx.x == 0 && threadIdx.x == 0)
        out[N_GRAPH * 2] = regsum[0] / ((float)N_EDGES * (float)(LLAYERS + 1));
}

// ---------------------------------------------------------------------------

extern "C" void kernel_launch(void* const* d_in, const int* in_sizes, int n_in,
                              void* d_out, int out_size, void* d_ws, size_t ws_size,
                              hipStream_t stream) {
    (void)in_sizes; (void)n_in; (void)out_size; (void)ws_size;

    const float* x      = (const float*)d_in[0];
    const float* w_in   = (const float*)d_in[1];
    const float* b_in   = (const float*)d_in[2];
    const float* ln_g   = (const float*)d_in[3];
    const float* ln_b   = (const float*)d_in[4];
    const float* cheb_w = (const float*)d_in[5];
    const float* cheb_b = (const float*)d_in[6];
    const float* bn_g   = (const float*)d_in[7];
    const float* bn_b   = (const float*)d_in[8];
    const float* w1     = (const float*)d_in[9];
    const float* b1     = (const float*)d_in[10];
    const float* w2     = (const float*)d_in[11];
    const float* b2     = (const float*)d_in[12];
    const i64*   eidx   = (const i64*)d_in[13];   // int64 per reference
    const i64*   batch  = (const i64*)d_in[14];
    const i64*   src    = eidx;
    const i64*   dst    = eidx + N_EDGES;
    float* out = (float*)d_out;

    // workspace carve-up (~103 MB)
    char* ws = (char*)d_ws;
    size_t off = 0;
    auto alloc = [&](size_t bytes) -> char* {
        char* p = ws + off;
        off = (off + bytes + 255) & ~(size_t)255;
        return p;
    };
    const size_t NH = (size_t)N_NODES * HDIM;
    float*    h      = (float*)alloc(NH * 4);
    float*    t1     = (float*)alloc(NH * 4);
    float*    s      = (float*)alloc(NH * 4);     // spmm(t1); T2 built on the fly
    float*    acc    = (float*)alloc(NH * 4);
    float*    deg    = (float*)alloc((size_t)N_NODES * 4);
    float*    dinv   = (float*)alloc((size_t)N_NODES * 4);
    _Float16* wt     = (_Float16*)alloc((size_t)(1 + LLAYERS * KORD) * HDIM * HDIM * 2);
    float*    sums   = (float*)alloc(2 * HDIM * 4);
    float*    ss     = (float*)alloc(2 * HDIM * 4);
    float*    counts = (float*)alloc((size_t)N_GRAPH * 4);
    float*    gsum   = (float*)alloc((size_t)N_GRAPH * HDIM * 4);
    float*    gmean  = (float*)alloc((size_t)N_GRAPH * HDIM * 4);
    float*    z      = (float*)alloc((size_t)N_GRAPH * 64 * 4);
    float*    regsum = (float*)alloc(16);

    const dim3 B(256);
    const int  GB   = (N_NODES + 127) / 128;                       // gemm blocks
    const int  EG   = (int)(((long long)N_EDGES * 32 + 255) / 256);// edge x 32-lane grids
    const int  NHB  = (int)((NH + 255) / 256);
    const int  MATS = HDIM * HDIM;

    // degree -> D^{-1/2}
    k_zero<<<(N_NODES + 255) / 256, B, 0, stream>>>(deg, N_NODES);
    k_deg <<<(N_EDGES + 255) / 256, B, 0, stream>>>(src, deg, N_EDGES);
    k_dinv<<<(N_NODES + 255) / 256, B, 0, stream>>>(deg, dinv, N_NODES);

    // f16 transposed weights: wt[0]=w_in, wt[1..9]=cheb_w (layer-contiguous)
    k_convw<<<(1 * MATS + 255) / 256, B, 0, stream>>>(w_in, wt, 1);
    k_convw<<<(LLAYERS * KORD * MATS + 255) / 256, B, 0, stream>>>(cheb_w, wt + MATS, LLAYERS * KORD);

    // node embed: Linear -> LayerNorm -> ReLU
    k_gemm128<<<GB, B, 0, stream>>>(x, wt, b_in, acc, N_NODES);
    k_ln_relu<<<(N_NODES + 7) / 8, B, 0, stream>>>(acc, ln_g, ln_b, h, N_NODES);

    k_zero<<<1, B, 0, stream>>>(regsum, 1);
    k_reg<<<EG, B, 0, stream>>>(src, dst, h, regsum, N_EDGES);   // hidden 0

    for (int l = 0; l < LLAYERS; ++l) {
        const _Float16* W012 = wt + (size_t)(1 + l * KORD) * MATS; // W0,W1,W2 contiguous
        const float*    cb   = cheb_b + l * HDIM;

        // T1 = L_hat h ; s = L_hat T1 (T2 = 2s - h built inside the GEMM)
        k_zero<<<NHB, B, 0, stream>>>(t1, (int)NH);
        k_spmm<<<EG, B, 0, stream>>>(src, dst, dinv, h, t1, N_EDGES);
        k_zero<<<NHB, B, 0, stream>>>(s, (int)NH);
        k_spmm<<<EG, B, 0, stream>>>(src, dst, dinv, t1, s, N_EDGES);

        // acc = T0 W0 + T1 W1 + T2 W2 + b   (one fused kernel, 96 WMMAs/wave)
        k_cheb_gemm<<<GB, B, 0, stream>>>(h, t1, s, W012, cb, acc, N_NODES);

        // BatchNorm (training stats) + ReLU -> h
        k_zero<<<1, B, 0, stream>>>(sums, 2 * HDIM);
        k_colstat<<<(N_NODES + 255) / 256, dim3(128), 0, stream>>>(acc, sums, N_NODES);
        k_bnfin<<<1, dim3(128), 0, stream>>>(sums, bn_g + l * HDIM, bn_b + l * HDIM, ss, N_NODES);
        k_bnapply<<<NHB, B, 0, stream>>>(acc, ss, h, (int)NH);

        k_reg<<<EG, B, 0, stream>>>(src, dst, h, regsum, N_EDGES); // hidden l+1
    }

    // global mean pool
    k_zero<<<1, B, 0, stream>>>(counts, N_GRAPH);
    k_zero<<<(N_GRAPH * HDIM + 255) / 256, B, 0, stream>>>(gsum, N_GRAPH * HDIM);
    k_counts<<<(N_NODES + 255) / 256, B, 0, stream>>>(batch, counts, N_NODES);
    k_gsum<<<(int)(((long long)N_NODES * 32 + 255) / 256), B, 0, stream>>>(batch, h, gsum, N_NODES);
    k_gmean<<<(N_GRAPH * HDIM + 255) / 256, B, 0, stream>>>(gsum, counts, gmean);

    // readout MLP -> logits, then reg scalar
    k_mlp1<<<(N_GRAPH * 64 + 255) / 256, B, 0, stream>>>(gmean, w1, b1, z);
    k_mlp2<<<1, B, 0, stream>>>(z, w2, b2, out);
    k_regfin<<<1, dim3(32), 0, stream>>>(regsum, out);
}
